// HierarchicalRNN_42786464203026
// MI455X (gfx1250) — matrix operand order repaired and checked by reading, not compiled, not run on hardware
//
#include <hip/hip_runtime.h>

// ---------------------------------------------------------------------------
// Hierarchical RNN scan on MI455X (gfx1250).
//
// Roofline: ~27 GFLOP vs ~770 MB of streamed noise -> HBM-bound (~33us floor
// at 23.3 TB/s).  One wave32 owns a 16-row batch tile for all T=1024 steps.
// State is kept transposed (r^T) so:
//   * weights live in A-layout VGPR fragments for the whole scan,
//   * WMMA B fragments are contiguous 32B LDS rows of the f16 state,
//   * WMMA D fragments store back as one 16B LDS write per fragment,
//   * noise loads for the C accumulator are contiguous 32B global loads.
// Pad columns (30->32, 60->64) carry garbage, which is inert because all
// weight fragments / win_r / wout_r are zero-masked there.  Only the final
// time step uses clamped (branchless) loads to avoid reading past the
// buffer ends.  14 x v_wmma_f32_16x16x32_f16 per step, f32 accumulation.
// ---------------------------------------------------------------------------

constexpr int cN1  = 30;
constexpr int cN2  = 60;
constexpr int cT   = 1024;
constexpr int cB   = 2048;
constexpr int TILE = 16;   // batch rows per wave == WMMA N dimension

typedef _Float16 v16h __attribute__((ext_vector_type(16)));
typedef _Float16 v8h  __attribute__((ext_vector_type(8)));
typedef float    v8f  __attribute__((ext_vector_type(8)));
typedef float    v4f  __attribute__((ext_vector_type(4)));

struct FalseT { static constexpr bool value = false; };
struct TrueT  { static constexpr bool value = true;  };

__device__ __forceinline__ v8f wmma16(v16h a, v16h b, v8f c) {
  // D = A(16x32 f16) * B(32x16 f16) + C(16x16 f32)
  return __builtin_amdgcn_wmma_f32_16x16x32_f16(false, a, false, b, (short)0, c,
                                                false, false);
}

__device__ __forceinline__ float fast_tanh(float x) {
#if __has_builtin(__builtin_amdgcn_tanhf)
  return __builtin_amdgcn_tanhf(x);          // native v_tanh_f32 on gfx1250
#elif __has_builtin(__builtin_amdgcn_tanh_f32)
  return __builtin_amdgcn_tanh_f32(x);
#else
  const float e = __expf(2.0f * x);          // v_exp_f32 fallback
  return 1.0f - 2.0f / (e + 1.0f);
#endif
}

// Unconditional 8-float load (32B).  May read a few bytes of the *next* row
// into pad elements -- harmless (pad weights are zero).
__device__ __forceinline__ void load8_fast(const float* __restrict__ p,
                                           float buf[8]) {
  v4f a, b;
  __builtin_memcpy(&a, p, 16);
  __builtin_memcpy(&b, p + 4, 16);
#pragma unroll
  for (int e = 0; e < 4; ++e) { buf[e] = a[e]; buf[4 + e] = b[e]; }
}

// Branchless clamped load for the final time step: addresses are clamped
// in-bounds (plain loads, no EXEC divergence), values masked with v_cndmask.
__device__ __forceinline__ void load8_safe(const float* __restrict__ row,
                                           int j0, int limit, float buf[8]) {
#pragma unroll
  for (int e = 0; e < 8; ++e) {
    const int j  = j0 + e;
    const int jc = (j < limit) ? j : (limit - 1);
    const float v = row[jc];
    buf[e] = (j < limit) ? v : 0.0f;
  }
}

__global__ __launch_bounds__(32) void hier_rnn_kernel(
    const float* __restrict__ inp, const float* __restrict__ noise1,
    const float* __restrict__ noise2, const float* __restrict__ W11,
    const float* __restrict__ W22, const float* __restrict__ W21,
    const float* __restrict__ W21_mask, const float* __restrict__ Win,
    const float* __restrict__ Wout, float* __restrict__ out) {
  __shared__ _Float16 s_r1[TILE][32];   // r1^T state tile [batch][neuron] f16
  __shared__ _Float16 s_r2[TILE][64];   // r2^T state tile

  const int lane = threadIdx.x;         // 0..31 (wave32)
  const int col  = lane & 15;           // batch row in tile == WMMA N index
  const int hi   = lane >> 4;           // wave half
  const int base = blockIdx.x * TILE;   // batch base of this tile

  // ---- A-layout weight fragments (VGPR-resident for the whole scan) ------
  // A-frag (16-bit, 16x32): lane -> M = col + 16*mtile,
  //                         elem e -> K = e + 16*hi
  // All loads use clamped in-bounds addresses + value masking: branchless.
  v16h a_w11[2];       // W11      : M=32, K=32
  v16h a_w21[4];       // W21*mask : M=64, K=32
  v16h a_w22[4][2];    // W22      : M=64, K=64 (2 K-chunks)
#pragma unroll
  for (int u = 0; u < 2; ++u) {
    const int M  = 16 * u + col;
    const int Mc = (M < cN1) ? M : (cN1 - 1);
#pragma unroll
    for (int e = 0; e < 16; ++e) {
      const int K  = e + (hi << 4);
      const int Kc = (K < cN1) ? K : (cN1 - 1);
      const float v = W11[Mc * cN1 + Kc];
      a_w11[u][e] = (_Float16)((M < cN1 && K < cN1) ? v : 0.0f);
    }
  }
#pragma unroll
  for (int u = 0; u < 4; ++u) {
    const int M  = 16 * u + col;
    const int Mc = (M < cN2) ? M : (cN2 - 1);
#pragma unroll
    for (int e = 0; e < 16; ++e) {
      const int K   = e + (hi << 4);
      const int Kc1 = (K < cN1) ? K : (cN1 - 1);
      const float w21v = W21[Mc * cN1 + Kc1] * W21_mask[Mc * cN1 + Kc1];
      a_w21[u][e] = (_Float16)((M < cN2 && K < cN1) ? w21v : 0.0f);
#pragma unroll
      for (int kc = 0; kc < 2; ++kc) {
        const int Kk  = 32 * kc + K;
        const int Kkc = (Kk < cN2) ? Kk : (cN2 - 1);
        const float v = W22[Mc * cN2 + Kkc];
        a_w22[u][kc][e] = (_Float16)((M < cN2 && Kk < cN2) ? v : 0.0f);
      }
    }
  }
  // D-layout per-lane constants: D-frag u, elem i -> M = 16u + 8*hi + i
  float win_r[16], wout_r[32];
#pragma unroll
  for (int u = 0; u < 4; ++u)
#pragma unroll
    for (int i = 0; i < 8; ++i) {
      const int M = 16 * u + 8 * hi + i;
      if (u < 2) {
        const float v = Win[(M < cN1) ? M : (cN1 - 1)];
        win_r[u * 8 + i] = (M < cN1) ? v : 0.0f;
      }
      const float v = Wout[(M < cN2) ? M : (cN2 - 1)];
      wout_r[u * 8 + i] = (M < cN2) ? v : 0.0f;
    }

  // r2_0 = 0 (s_r1 is fully rewritten before it is ever read)
  {
    _Float16* z2 = &s_r2[0][0];
    for (int i = lane; i < TILE * 64; i += 32) z2[i] = (_Float16)0.0f;
  }
  __syncthreads();

  v16h b_r1old = {};                    // r1_0 = 0, carried in registers
  const int colN1 = col * cN1;
  const int colN2 = col * cN2;

  // One RNN step.  SAFE=true only for t = T-1 (clamped loads).
  auto step = [&](const float* __restrict__ n1b, const float* __restrict__ n2b,
                  const float* __restrict__ xb, float* __restrict__ yb,
                  auto safe_tag) {
    constexpr bool SAFE = decltype(safe_tag)::value;
    const float x = xb[col];            // each lane owns one batch column
    if constexpr (!SAFE) {              // prefetch next step's noise rows
      __builtin_prefetch(n1b + colN1 + (size_t)cB * cN1, 0, 0);
      __builtin_prefetch(n2b + colN2 + (size_t)cB * cN2, 0, 0);
    }
    // old r2 state as B fragments (prev-iter stores fenced by loop-end sync)
    v16h b_r2lo, b_r2hi;
    __builtin_memcpy(&b_r2lo, &s_r2[col][16 * hi], 32);
    __builtin_memcpy(&b_r2hi, &s_r2[col][32 + 16 * hi], 32);

    // ---- layer 1: r1' = tanh(W11 * r1^T + Win * x^T + n1^T) ----
    v8f d1[2];
#pragma unroll
    for (int u = 0; u < 2; ++u) {
      float buf[8];
      const int j0 = 16 * u + 8 * hi;
      if constexpr (SAFE) load8_safe(n1b + colN1, j0, cN1, buf);
      else                load8_fast(n1b + colN1 + j0, buf);
      v8f c;
#pragma unroll
      for (int i = 0; i < 8; ++i) c[i] = buf[i] + win_r[u * 8 + i] * x;
      d1[u] = wmma16(a_w11[u], b_r1old, c);
    }
#pragma unroll
    for (int u = 0; u < 2; ++u) {       // tanh -> f16 -> LDS (one 16B store)
      v8h h;
#pragma unroll
      for (int i = 0; i < 8; ++i) h[i] = (_Float16)fast_tanh(d1[u][i]);
      *(v8h*)&s_r1[col][16 * u + 8 * hi] = h;
    }
    __syncthreads();                    // r1' visible across lane halves
    v16h b_r1n;
    __builtin_memcpy(&b_r1n, &s_r1[col][16 * hi], 32);

    // ---- layer 2: r2' = tanh(W22 * r2^T + W21m * r1'^T + n2^T) ----
    v8f d2[4];
#pragma unroll
    for (int u = 0; u < 4; ++u) {
      float buf[8];
      const int j0 = 16 * u + 8 * hi;
      if constexpr (SAFE) load8_safe(n2b + colN2, j0, cN2, buf);
      else                load8_fast(n2b + colN2 + j0, buf);
      v8f c;
#pragma unroll
      for (int i = 0; i < 8; ++i) c[i] = buf[i];
      c     = wmma16(a_w21[u],    b_r1n,  c);
      c     = wmma16(a_w22[u][0], b_r2lo, c);
      d2[u] = wmma16(a_w22[u][1], b_r2hi, c);
    }
    float part = 0.0f;
#pragma unroll
    for (int u = 0; u < 4; ++u) {
      v8h h;
#pragma unroll
      for (int i = 0; i < 8; ++i) {
        const float tv = fast_tanh(d2[u][i]);
        part += wout_r[u * 8 + i] * tv;           // y partial dot
        h[i] = (_Float16)tv;
      }
      *(v8h*)&s_r2[col][16 * u + 8 * hi] = h;     // one 16B store
    }
    part += __shfl_xor(part, 16, 32);   // combine disjoint neuron halves
    if (hi == 0) yb[col] = part;        // coalesced 64B store
    b_r1old = b_r1n;                    // carry r1 fragment in registers
    __syncthreads();                    // r2' visible before next iteration
  };

  // uniform scalar bases advance in SGPRs; lane offsets are small 32-bit
  const float* n1b = noise1 + (size_t)base * cN1;
  const float* n2b = noise2 + (size_t)base * cN2;
  const float* xb  = inp + base;
  float*       yb  = out + base;
  for (int t = 0; t < cT - 1; ++t) {
    step(n1b, n2b, xb, yb, FalseT{});
    n1b += (size_t)cB * cN1;
    n2b += (size_t)cB * cN2;
    xb  += cB;
    yb  += cB;
  }
  step(n1b, n2b, xb, yb, TrueT{});      // final step: clamped loads
}

extern "C" void kernel_launch(void* const* d_in, const int* in_sizes, int n_in,
                              void* d_out, int out_size, void* d_ws,
                              size_t ws_size, hipStream_t stream) {
  (void)in_sizes; (void)n_in; (void)out_size; (void)d_ws; (void)ws_size;
  const float* inp      = (const float*)d_in[0];
  const float* noise1   = (const float*)d_in[1];
  const float* noise2   = (const float*)d_in[2];
  const float* W11      = (const float*)d_in[3];
  const float* W22      = (const float*)d_in[4];
  const float* W21      = (const float*)d_in[5];
  const float* W21_mask = (const float*)d_in[6];
  const float* Win      = (const float*)d_in[7];
  const float* Wout     = (const float*)d_in[8];
  float*       out      = (float*)d_out;

  hier_rnn_kernel<<<cB / TILE, 32, 0, stream>>>(inp, noise1, noise2, W11, W22,
                                                W21, W21_mask, Win, Wout, out);
}
